// Attention_54425825575529
// MI455X (gfx1250) — 8-bit matrix operand order repaired and checked by reading, compiled, not run on hardware
//
#include <hip/hip_runtime.h>


typedef int v8i __attribute__((ext_vector_type(8)));

union U32x8 { v8i v; unsigned u[8]; };

__device__ __forceinline__ unsigned encf(float f) {
    unsigned u = __float_as_uint(f);
    return (u & 0x80000000u) ? ~u : (u | 0x80000000u);
}
__device__ __forceinline__ float decf(unsigned e) {
    unsigned u = (e & 0x80000000u) ? (e ^ 0x80000000u) : ~e;
    return __uint_as_float(u);
}

__device__ __forceinline__ int kbaseA(int v, int half) {
    return ((v >> 1) << 4) + ((v & 1) << 2) + half * 8;
}
__device__ __forceinline__ int kbaseB(int v, int half) {
    return ((v >> 1) << 4) + ((v & 1) << 2) + half * 8;
}

__device__ __forceinline__ void async_ld_b128(unsigned ldsOff,
                                              const void* gaddr) {
    asm volatile("global_load_async_to_lds_b128 %0, %1, off"
                 :: "v"(ldsOff), "v"(gaddr) : "memory");
}
__device__ __forceinline__ void wait_async_le3() {
    asm volatile("s_wait_asynccnt 0x3" ::: "memory");
}
__device__ __forceinline__ void wait_async_0() {
    asm volatile("s_wait_asynccnt 0x0" ::: "memory");
}

__global__ void __launch_bounds__(256)
minmax_f32(const float* __restrict__ src, size_t n,
           unsigned* mnSlot, unsigned* mxSlot) {
    float lo = 3.4e38f, hi = -3.4e38f;
    for (size_t i = (size_t)blockIdx.x * blockDim.x + threadIdx.x; i < n;
         i += (size_t)gridDim.x * blockDim.x) {
        float v = src[i];
        lo = fminf(lo, v);
        hi = fmaxf(hi, v);
    }
    __shared__ float slo[256], shi[256];
    slo[threadIdx.x] = lo; shi[threadIdx.x] = hi;
    __syncthreads();
    for (int off = 128; off > 0; off >>= 1) {
        if ((int)threadIdx.x < off) {
            slo[threadIdx.x] = fminf(slo[threadIdx.x], slo[threadIdx.x + off]);
            shi[threadIdx.x] = fmaxf(shi[threadIdx.x], shi[threadIdx.x + off]);
        }
        __syncthreads();
    }
    if (threadIdx.x == 0) {
        atomicMin(mnSlot, encf(slo[0]));
        atomicMax(mxSlot, encf(shi[0]));
    }
}

__global__ void qparams_k(const unsigned* mn, const unsigned* mx, float* sz) {
    if (threadIdx.x == 0 && blockIdx.x == 0) {
        float lo = decf(*mn), hi = decf(*mx);
        float s = (hi - lo) * (1.0f / 15.0f);
        float z = rintf(15.0f - hi / s);
        sz[0] = s; sz[1] = z;
    }
}

__global__ void __launch_bounds__(256)
quantize_rows(const float* __restrict__ src, signed char* __restrict__ dstQ,
              float* __restrict__ rowsum, const float* __restrict__ sz, int C) {
    const int r = blockIdx.x;
    const float inv = 1.0f / sz[0];
    const float z = sz[1];
    float part = 0.0f;
    for (int c = threadIdx.x; c < C; c += 256) {
        float q = rintf(src[(size_t)r * C + c] * inv + z);
        dstQ[(size_t)r * C + c] = (signed char)q;
        part += q;
    }
    __shared__ float red[256];
    red[threadIdx.x] = part;
    __syncthreads();
    for (int off = 128; off > 0; off >>= 1) {
        if ((int)threadIdx.x < off) red[threadIdx.x] += red[threadIdx.x + off];
        __syncthreads();
    }
    if (threadIdx.x == 0) rowsum[r] = red[0];
}

__global__ void __launch_bounds__(256)
row_sum_f32(const float* __restrict__ src, float* __restrict__ dst, int C) {
    const int r = blockIdx.x;
    float part = 0.0f;
    for (int c = threadIdx.x; c < C; c += 256)
        part += src[(size_t)r * C + c];
    __shared__ float red[256];
    red[threadIdx.x] = part;
    __syncthreads();
    for (int off = 128; off > 0; off >>= 1) {
        if ((int)threadIdx.x < off) red[threadIdx.x] += red[threadIdx.x + off];
        __syncthreads();
    }
    if (threadIdx.x == 0) dst[r] = red[0];
}

__global__ void __launch_bounds__(256)
gemm_iu8(const signed char* __restrict__ A, const signed char* __restrict__ Bn,
         float* __restrict__ Cout,
         const float* __restrict__ colsumB, const float* __restrict__ rowsumA,
         const float* __restrict__ pA, const float* __restrict__ pB,
         float kcorr, const float* __restrict__ bias,
         unsigned* mmMin, unsigned* mmMax,
         int M, int N, int K) {
    __shared__ alignas(16) signed char ldsA[2][64 * 64];
    __shared__ alignas(16) signed char ldsB[2][128 * 64];
    const int tid  = threadIdx.x;
    const int lane = tid & 31;
    const int w    = tid >> 5;
    const int half = lane >> 4;
    const int l16  = lane & 15;
    const int mb = blockIdx.y * 64, nb = blockIdx.x * 128;
    const int wm = (w >> 1) * 16,  wn = (w & 1) * 64;

    const int ar = tid >> 2, ac = (tid & 3) << 4;
    const int br = tid >> 1, bc = (tid & 1) << 5;
    const signed char* gA = A + (size_t)(mb + ar) * K + ac;
    const signed char* gB = Bn + (size_t)(nb + br) * K + bc;
    const unsigned aOff = (unsigned)(size_t)(&ldsA[0][0]) + (unsigned)(ar * 64 + ac);
    const unsigned bOff = (unsigned)(size_t)(&ldsB[0][0]) + (unsigned)(br * 64 + bc);

    async_ld_b128(aOff, gA);
    async_ld_b128(bOff, gB);
    async_ld_b128(bOff + 16, gB + 16);

    v8i acc[4] = {};
    const int nsteps = K >> 6;
    for (int s = 0; s < nsteps; ++s) {
        const int cur = s & 1;
        if (s + 1 < nsteps) {
            const int k0 = (s + 1) << 6;
            const unsigned bsel = (unsigned)((s + 1) & 1);
            async_ld_b128(aOff + bsel * 4096, gA + k0);
            async_ld_b128(bOff + bsel * 8192, gB + k0);
            async_ld_b128(bOff + bsel * 8192 + 16, gB + k0 + 16);
            wait_async_le3();
        } else {
            wait_async_0();
        }
        __syncthreads();

        U32x8 a;
        const signed char* la = &ldsA[cur][(wm + l16) * 64];
#pragma unroll
        for (int v = 0; v < 8; ++v)
            a.u[v] = *(const unsigned*)(la + kbaseA(v, half));
#pragma unroll
        for (int t = 0; t < 4; ++t) {
            const signed char* lb = &ldsB[cur][(wn + t * 16 + l16) * 64];
            U32x8 b;
#pragma unroll
            for (int v = 0; v < 8; ++v)
                b.u[v] = *(const unsigned*)(lb + kbaseB(v, half));
            acc[t] = __builtin_amdgcn_wmma_i32_16x16x64_iu8(
                false, a.v, false, b.v, acc[t], false, false);
        }
        __syncthreads();
    }

    const float sA = pA[0], zA = pA[1], sB = pB[0], zB = pB[1];
    const float sc = sA * sB;
    const float zz = zA * zB * kcorr;
    float lmin = 3.4e38f, lmax = -3.4e38f;
#pragma unroll
    for (int t = 0; t < 4; ++t) {
        const int c = nb + wn + t * 16 + l16;
        const float cb = colsumB[c];
        const float bi = bias ? bias[c] : 0.0f;
#pragma unroll
        for (int v = 0; v < 8; ++v) {
            const int r = mb + wm + v + half * 8;
            float val = ((float)acc[t][v] - cb * zA - rowsumA[r] * zB + zz) * sc + bi;
            Cout[(size_t)r * N + c] = val;
            lmin = fminf(lmin, val);
            lmax = fmaxf(lmax, val);
        }
    }
    if (mmMin) {
        for (int d = 16; d; d >>= 1) {
            lmin = fminf(lmin, __shfl_xor(lmin, d));
            lmax = fmaxf(lmax, __shfl_xor(lmax, d));
        }
        if (lane == 0) {
            const int g = (nb + wn) / 768;
            atomicMin(&mmMin[g], encf(lmin));
            atomicMax(&mmMax[g], encf(lmax));
        }
    }
}

__global__ void __launch_bounds__(64)
quantize_qkv(const float* __restrict__ xqkv,
             signed char* __restrict__ qq, signed char* __restrict__ kk,
             signed char* __restrict__ vqT,
             float* __restrict__ rsq, float* __restrict__ csk,
             float* __restrict__ csv,
             const float* pQ, const float* pK, const float* pV) {
    const int idx = blockIdx.x;
    const int bh = idx >> 10;
    const int n  = idx & 1023;
    const int b  = bh / 12;
    const int h  = bh - b * 12;
    const int d  = threadIdx.x;
    const float* row = xqkv + (size_t)(b * 1024 + n) * 2304;
    const int col = h * 64 + d;

    const float qv = rintf(row[col]         * (1.0f / pQ[0]) + pQ[1]);
    const float kv = rintf(row[768  + col]  * (1.0f / pK[0]) + pK[1]);
    const float vv = rintf(row[1536 + col]  * (1.0f / pV[0]) + pV[1]);

    qq[((size_t)bh * 1024 + n) * 64 + d] = (signed char)qv;
    kk[((size_t)bh * 1024 + n) * 64 + d] = (signed char)kv;
    vqT[((size_t)bh * 64 + d) * 1024 + n] = (signed char)vv;
    atomicAdd(&csv[bh * 64 + d], vv);

    __shared__ float s1[64], s2[64];
    s1[d] = qv; s2[d] = kv;
    __syncthreads();
    for (int off = 32; off > 0; off >>= 1) {
        if (d < off) { s1[d] += s1[d + off]; s2[d] += s2[d + off]; }
        __syncthreads();
    }
    if (d == 0) {
        rsq[(size_t)bh * 1024 + n] = s1[0];
        csk[(size_t)bh * 1024 + n] = s2[0];
    }
}

__global__ void __launch_bounds__(128)
attn_pass1(const signed char* __restrict__ qq, const signed char* __restrict__ kk,
           const float* __restrict__ rsq, const float* __restrict__ csk,
           const float* pQ, const float* pK,
           float* __restrict__ mArr, float* __restrict__ lArr,
           float* __restrict__ mnArr) {
    const int w = threadIdx.x >> 5, lane = threadIdx.x & 31;
    const int half = lane >> 4, l16 = lane & 15;
    const int gw = blockIdx.x * 4 + w;
    const int bh = gw >> 6;
    const int row0 = (gw & 63) * 16;
    const signed char* qbase = qq + (size_t)bh * 1024 * 64;
    const signed char* kbase = kk + (size_t)bh * 1024 * 64;
    const float zq = pQ[1], zk = pK[1];
    const float scl = 0.125f * pQ[0] * pK[0];

    U32x8 a;
    {
        const signed char* ar = qbase + (size_t)(row0 + l16) * 64;
#pragma unroll
        for (int v = 0; v < 8; ++v)
            a.u[v] = *(const unsigned*)(ar + kbaseA(v, half));
    }
    float rs8[8], m8[8], l8[8], mn8[8];
#pragma unroll
    for (int v = 0; v < 8; ++v) {
        rs8[v] = rsq[(size_t)bh * 1024 + row0 + v + half * 8];
        m8[v] = -1e30f; l8[v] = 0.0f; mn8[v] = 1e30f;
    }

    for (int j0 = 0; j0 < 1024; j0 += 16) {
        U32x8 bfr;
        const signed char* br = kbase + (size_t)(j0 + l16) * 64;
#pragma unroll
        for (int v = 0; v < 8; ++v)
            bfr.u[v] = *(const unsigned*)(br + kbaseB(v, half));
        v8i dd = {};
        dd = __builtin_amdgcn_wmma_i32_16x16x64_iu8(false, a.v, false, bfr.v,
                                                    dd, false, false);
        const float cs = csk[(size_t)bh * 1024 + j0 + l16];
#pragma unroll
        for (int v = 0; v < 8; ++v) {
            float f = ((float)dd[v] - cs * zq - rs8[v] * zk) * scl;
            float nm = fmaxf(m8[v], f);
            l8[v] = l8[v] * __expf(m8[v] - nm) + __expf(f - nm);
            m8[v] = nm;
            mn8[v] = fminf(mn8[v], f);
        }
    }
    for (int d = 1; d < 16; d <<= 1) {
#pragma unroll
        for (int v = 0; v < 8; ++v) {
            float mo = __shfl_xor(m8[v], d);
            float lo = __shfl_xor(l8[v], d);
            float no = __shfl_xor(mn8[v], d);
            float nm = fmaxf(m8[v], mo);
            l8[v] = l8[v] * __expf(m8[v] - nm) + lo * __expf(mo - nm);
            m8[v] = nm;
            mn8[v] = fminf(mn8[v], no);
        }
    }
    if (l16 == 0) {
#pragma unroll
        for (int v = 0; v < 8; ++v) {
            const size_t r = (size_t)bh * 1024 + row0 + v + half * 8;
            mArr[r] = m8[v]; lArr[r] = l8[v]; mnArr[r] = mn8[v];
        }
    }
}

__global__ void __launch_bounds__(256)
a_minmax(const float* __restrict__ mA, const float* __restrict__ lA,
         const float* __restrict__ mnA, int n,
         unsigned* mnSlot, unsigned* mxSlot) {
    float lo = 3.4e38f, hi = -3.4e38f;
    for (int i = blockIdx.x * blockDim.x + threadIdx.x; i < n;
         i += gridDim.x * blockDim.x) {
        const float l = lA[i];
        hi = fmaxf(hi, 1.0f / l);
        lo = fminf(lo, __expf(mnA[i] - mA[i]) / l);
    }
    atomicMin(mnSlot, encf(lo));
    atomicMax(mxSlot, encf(hi));
}

__global__ void __launch_bounds__(128)
attn_pass2(const signed char* __restrict__ qq, const signed char* __restrict__ kk,
           const signed char* __restrict__ vqT,
           const float* __restrict__ rsq, const float* __restrict__ csk,
           const float* __restrict__ csv,
           const float* __restrict__ mArr, const float* __restrict__ lArr,
           const float* pQ, const float* pK, const float* pA, const float* pV,
           float* __restrict__ outF) {
    __shared__ alignas(16) signed char atile[4][16 * 64];
    const int w = threadIdx.x >> 5, lane = threadIdx.x & 31;
    const int half = lane >> 4, l16 = lane & 15;
    const int gw = blockIdx.x * 4 + w;
    const int bh = gw >> 6;
    const int row0 = (gw & 63) * 16;
    const int b = bh / 12;
    const int h = bh - b * 12;
    const signed char* qbase = qq + (size_t)bh * 1024 * 64;
    const signed char* kbase = kk + (size_t)bh * 1024 * 64;
    const float zq = pQ[1], zk = pK[1];
    const float scl = 0.125f * pQ[0] * pK[0];
    const float sa = pA[0], za = pA[1];
    const float sv = pV[0], zv = pV[1];
    const float inv_sa = 1.0f / sa;

    U32x8 a;
    {
        const signed char* ar = qbase + (size_t)(row0 + l16) * 64;
#pragma unroll
        for (int v = 0; v < 8; ++v)
            a.u[v] = *(const unsigned*)(ar + kbaseA(v, half));
    }
    float rs8[8], m8[8], il8[8], rsa8[8] = {};
#pragma unroll
    for (int v = 0; v < 8; ++v) {
        const size_t r = (size_t)bh * 1024 + row0 + v + half * 8;
        rs8[v] = rsq[r];
        m8[v] = mArr[r];
        il8[v] = 1.0f / lArr[r];
    }
    v8i oacc[4] = {};
    signed char* myTile = &atile[w][0];

    for (int kc = 0; kc < 1024; kc += 64) {
#pragma unroll
        for (int s = 0; s < 4; ++s) {
            const int j0 = kc + s * 16;
            U32x8 bfr;
            const signed char* br = kbase + (size_t)(j0 + l16) * 64;
#pragma unroll
            for (int v = 0; v < 8; ++v)
                bfr.u[v] = *(const unsigned*)(br + kbaseB(v, half));
            v8i dd = {};
            dd = __builtin_amdgcn_wmma_i32_16x16x64_iu8(false, a.v, false, bfr.v,
                                                        dd, false, false);
            const float cs = csk[(size_t)bh * 1024 + j0 + l16];
#pragma unroll
            for (int v = 0; v < 8; ++v) {
                float f = ((float)dd[v] - cs * zq - rs8[v] * zk) * scl;
                float av = __expf(f - m8[v]) * il8[v];
                float qa = rintf(av * inv_sa + za);
                rsa8[v] += qa;
                myTile[(v + half * 8) * 64 + s * 16 + l16] = (signed char)qa;
            }
        }
        __syncthreads();
        U32x8 af;
#pragma unroll
        for (int v = 0; v < 8; ++v)
            af.u[v] = *(const unsigned*)(myTile + l16 * 64 + kbaseA(v, half));
#pragma unroll
        for (int t = 0; t < 4; ++t) {
            const signed char* vb =
                vqT + ((size_t)bh * 64 + t * 16 + l16) * 1024 + kc;
            U32x8 bf;
#pragma unroll
            for (int v = 0; v < 8; ++v)
                bf.u[v] = *(const unsigned*)(vb + kbaseB(v, half));
            oacc[t] = __builtin_amdgcn_wmma_i32_16x16x64_iu8(
                false, af.v, false, bf.v, oacc[t], false, false);
        }
        __syncthreads();
    }
    for (int d = 1; d < 16; d <<= 1) {
#pragma unroll
        for (int v = 0; v < 8; ++v) rsa8[v] += __shfl_xor(rsa8[v], d);
    }
    const float zz = za * zv * 1024.0f;
    const float sc2 = sa * sv;
#pragma unroll
    for (int t = 0; t < 4; ++t) {
        const int dcol = t * 16 + l16;
        const float cv = csv[bh * 64 + dcol];
#pragma unroll
        for (int v = 0; v < 8; ++v) {
            const int r = row0 + v + half * 8;
            float val = ((float)oacc[t][v] - cv * za - rsa8[v] * zv + zz) * sc2;
            outF[((size_t)(b * 1024 + r)) * 768 + h * 64 + dcol] = val;
        }
    }
}

extern "C" void kernel_launch(void* const* d_in, const int* in_sizes, int n_in,
                              void* d_out, int out_size, void* d_ws,
                              size_t ws_size, hipStream_t stream) {
    (void)in_sizes; (void)n_in; (void)out_size; (void)ws_size;
    const float* x     = (const float*)d_in[0];
    const float* w_qkv = (const float*)d_in[1];
    const float* w_out = (const float*)d_in[2];
    const float* b_out = (const float*)d_in[3];
    float* out = (float*)d_out;
    char* ws = (char*)d_ws;

    const int Mx = 4096, Dk = 768, N3 = 2304, BH = 48, SEQ = 1024, HD = 64;

    unsigned* mnS = (unsigned*)(ws + 0);
    unsigned* mxS = (unsigned*)(ws + 32);
    float* prm = (float*)(ws + 64);
    size_t cur = 256;
    auto alloc = [&](size_t bytes) -> char* {
        char* p = ws + cur;
        cur += (bytes + 255) & ~(size_t)255;
        return p;
    };
    signed char* xq   = (signed char*)alloc((size_t)Mx * Dk);
    signed char* w1q  = (signed char*)alloc((size_t)N3 * Dk);
    signed char* w2q  = (signed char*)alloc((size_t)Dk * Dk);
    float* rowsum_xq  = (float*)alloc(Mx * 4);
    float* colsum_w1  = (float*)alloc(N3 * 4);
    float* colsum_w2  = (float*)alloc(Dk * 4);
    float* xqkv       = (float*)alloc((size_t)Mx * N3 * 4);
    float* rowsum_xqkv= (float*)alloc(Mx * 4);
    signed char* qqB  = (signed char*)alloc((size_t)BH * SEQ * HD);
    signed char* kkB  = (signed char*)alloc((size_t)BH * SEQ * HD);
    signed char* vqT  = (signed char*)alloc((size_t)BH * HD * SEQ);
    float* rsq        = (float*)alloc((size_t)BH * SEQ * 4);
    float* csk        = (float*)alloc((size_t)BH * SEQ * 4);
    float* csv        = (float*)alloc((size_t)BH * HD * 4);
    float* mArr       = (float*)alloc((size_t)BH * SEQ * 4);
    float* lArr       = (float*)alloc((size_t)BH * SEQ * 4);
    float* mnArr      = (float*)alloc((size_t)BH * SEQ * 4);
    float* attnout    = (float*)alloc((size_t)Mx * Dk * 4);
    signed char* oq   = (signed char*)alloc((size_t)Mx * Dk);
    float* rowsum_oq  = (float*)alloc(Mx * 4);

    hipMemsetAsync(mnS, 0xFF, 32, stream);
    hipMemsetAsync(mxS, 0x00, 32, stream);
    hipMemsetAsync(csv, 0x00, (size_t)BH * HD * 4, stream);

    minmax_f32<<<512, 256, 0, stream>>>(x, (size_t)Mx * Dk, mnS + 0, mxS + 0);
    minmax_f32<<<512, 256, 0, stream>>>(w_qkv, (size_t)N3 * Dk, mnS + 1, mxS + 1);
    minmax_f32<<<256, 256, 0, stream>>>(w_out, (size_t)Dk * Dk, mnS + 2, mxS + 2);
    qparams_k<<<1, 1, 0, stream>>>(mnS + 0, mxS + 0, prm + 0);
    qparams_k<<<1, 1, 0, stream>>>(mnS + 1, mxS + 1, prm + 2);
    qparams_k<<<1, 1, 0, stream>>>(mnS + 2, mxS + 2, prm + 4);
    quantize_rows<<<Mx, 256, 0, stream>>>(x, xq, rowsum_xq, prm + 0, Dk);
    quantize_rows<<<N3, 256, 0, stream>>>(w_qkv, w1q, colsum_w1, prm + 2, Dk);
    quantize_rows<<<Dk, 256, 0, stream>>>(w_out, w2q, colsum_w2, prm + 4, Dk);

    gemm_iu8<<<dim3(N3 / 128, Mx / 64), 256, 0, stream>>>(
        xq, w1q, xqkv, colsum_w1, rowsum_xq, prm + 0, prm + 2,
        768.0f, nullptr, mnS + 3, mxS + 3, Mx, N3, Dk);
    row_sum_f32<<<Mx, 256, 0, stream>>>(xqkv, rowsum_xqkv, N3);
    qparams_k<<<1, 1, 0, stream>>>(mnS + 3, mxS + 3, prm + 6);
    qparams_k<<<1, 1, 0, stream>>>(mnS + 4, mxS + 4, prm + 8);
    qparams_k<<<1, 1, 0, stream>>>(mnS + 5, mxS + 5, prm + 10);
    quantize_qkv<<<BH * SEQ, 64, 0, stream>>>(xqkv, qqB, kkB, vqT, rsq, csk,
                                              csv, prm + 6, prm + 8, prm + 10);

    attn_pass1<<<BH * SEQ / 64, 128, 0, stream>>>(qqB, kkB, rsq, csk,
                                                  prm + 6, prm + 8,
                                                  mArr, lArr, mnArr);
    a_minmax<<<192, 256, 0, stream>>>(mArr, lArr, mnArr, BH * SEQ,
                                      mnS + 6, mxS + 6);
    qparams_k<<<1, 1, 0, stream>>>(mnS + 6, mxS + 6, prm + 12);
    attn_pass2<<<BH * SEQ / 64, 128, 0, stream>>>(
        qqB, kkB, vqT, rsq, csk, csv, mArr, lArr,
        prm + 6, prm + 8, prm + 12, prm + 10, attnout);

    minmax_f32<<<512, 256, 0, stream>>>(attnout, (size_t)Mx * Dk, mnS + 7, mxS + 7);
    qparams_k<<<1, 1, 0, stream>>>(mnS + 7, mxS + 7, prm + 14);
    quantize_rows<<<Mx, 256, 0, stream>>>(attnout, oq, rowsum_oq, prm + 14, Dk);
    gemm_iu8<<<dim3(Dk / 128, Mx / 64), 256, 0, stream>>>(
        oq, w2q, out, colsum_w2, rowsum_xqkv, prm + 14, prm + 4,
        2304.0f, b_out, nullptr, nullptr, Mx, Dk, Dk);
    (void)rowsum_oq;
}
